// ExtractorModel_46256797778325
// MI455X (gfx1250) — matrix operand order repaired and checked by reading, not compiled, not run on hardware
//
#include <hip/hip_runtime.h>
#include <math.h>

// ---------------------------------------------------------------- constants
#define Bb    64
#define Ss    512
#define Hh    768
#define G4    3072          // 4*H
#define TAGn  100
#define TAGP  128           // padded tag/att slot width
#define ROLEn 36
#define KR1   896           // 768 + 128   (stage-1 recurrent K)
#define KR2   1024          // 768 + 256   (stage-2 recurrent K)
#define RLD   1024          // r row stride (elements)
#define NBLK  96            // persistent grid: 96 blocks * 8 waves = 768 tiles
#define TPB   256

typedef unsigned short ushort_t;
typedef __attribute__((ext_vector_type(8)))  float    v8f;
typedef __attribute__((ext_vector_type(16))) __bf16   v16bf;
typedef __attribute__((ext_vector_type(8)))  ushort_t v8us;
typedef __attribute__((ext_vector_type(16))) ushort_t v16us;

// ---------------------------------------------------------------- helpers
__device__ __forceinline__ ushort_t f2bf(float f) {
  unsigned u = __builtin_bit_cast(unsigned, f);
  u += 0x7FFFu + ((u >> 16) & 1u);          // round-to-nearest-even
  return (ushort_t)(u >> 16);
}

__device__ __forceinline__ float sigmoidf_(float x) {
  return 1.0f / (1.0f + expf(-x));
}

// A fragment (16x32 bf16, M x K): lane holds row m0+(lane&15);
// elements 0..15 = K k0+g*16..k0+g*16+15 contiguous  (g = lane>>4)
__device__ __forceinline__ v16bf load_a_frag(const ushort_t* A, int lda,
                                             int m0, int k0, int lane) {
  const ushort_t* p = A + (size_t)(m0 + (lane & 15)) * lda + k0 + ((lane >> 4) << 4);
  v8us lo = *(const v8us*)p;
  v8us hi = *(const v8us*)(p + 8);
  v16us a;
#pragma unroll
  for (int i = 0; i < 8; ++i) { a[i] = lo[i]; a[i + 8] = hi[i]; }
  return __builtin_bit_cast(v16bf, a);
}

// B fragment (32x16 bf16, K x N) from W stored [N, K] row-major (B = W^T):
// lane holds column n0+(lane&15); 16 contiguous K values at k0+(lane>>4)*16
__device__ __forceinline__ v16bf load_b_frag(const ushort_t* W, int ldb,
                                             int n0, int k0, int lane) {
  const ushort_t* p = W + (size_t)(n0 + (lane & 15)) * ldb + k0 + ((lane >> 4) << 4);
  return __builtin_bit_cast(v16bf, *(const v16us*)p);
}

__device__ __forceinline__ v8f wmma_bf16(v16bf a, v16bf b, v8f c) {
  return __builtin_amdgcn_wmma_f32_16x16x32_bf16(false, a, false, b,
                                                 (short)0, c, false, false);
}

// grid-wide arrive/wait barrier (monotone counter, zeroed per launch)
__device__ __forceinline__ void grid_bar(unsigned* cnt, unsigned gen, unsigned nb) {
  __syncthreads();
  if (threadIdx.x == 0) {
    __threadfence();
    __hip_atomic_fetch_add(cnt, 1u, __ATOMIC_RELEASE, __HIP_MEMORY_SCOPE_AGENT);
    unsigned target = gen * nb;
    while (__hip_atomic_load(cnt, __ATOMIC_ACQUIRE, __HIP_MEMORY_SCOPE_AGENT) < target)
      __builtin_amdgcn_s_sleep(1);
  }
  __syncthreads();
  __threadfence();
}

// stage this wave's 16-row B-slice of the recurrent weights into LDS (once per launch)
__device__ __forceinline__ void stage_wr_to_lds(ushort_t* lds_slice,
                                                const ushort_t* Wr, int Kr,
                                                int n0, int lane) {
  const ushort_t* src = Wr + (size_t)n0 * Kr;
  const int total = 16 * Kr;                 // elements in slice
#pragma unroll 4
  for (int i = lane * 8; i < total; i += 32 * 8)
    *(v8us*)(lds_slice + i) = *(const v8us*)(src + i);
}

// ---------------------------------------------------------------- pack kernels
__global__ void pack_x_kernel(const float* __restrict__ X, ushort_t* __restrict__ Xbf) {
  // X: [B,S,H] f32 -> Xbf: [S,B,H] bf16
  size_t i = (size_t)blockIdx.x * blockDim.x + threadIdx.x;
  size_t total = (size_t)Ss * Bb * Hh;
  if (i >= total) return;
  int h = (int)(i % Hh);
  size_t t = i / Hh;
  int b = (int)(t % Bb);
  int s = (int)(t / Bb);
  Xbf[i] = f2bf(X[((size_t)b * Ss + s) * Hh + h]);
}

__global__ void pack_wr_kernel(const float* __restrict__ Whh, const float* __restrict__ Wih,
                               ushort_t* __restrict__ Wr, int Kr, int ih_ld) {
  // Wr [3072][Kr]: cols 0..767 = Whh[n][c]; col 768+128*j+t = (t<100) ? Wih[n][768+100*j+t] : 0
  size_t i = (size_t)blockIdx.x * blockDim.x + threadIdx.x;
  size_t total = (size_t)G4 * Kr;
  if (i >= total) return;
  int c = (int)(i % Kr);
  int n = (int)(i / Kr);
  float v;
  if (c < Hh) {
    v = Whh[(size_t)n * Hh + c];
  } else {
    int j = (c - Hh) >> 7;
    int t = (c - Hh) & 127;
    v = (t < TAGn) ? Wih[(size_t)n * ih_ld + Hh + TAGn * j + t] : 0.0f;
  }
  Wr[i] = f2bf(v);
}

__global__ void pack_wx_kernel(const float* __restrict__ Wih, ushort_t* __restrict__ Wx, int ih_ld) {
  size_t i = (size_t)blockIdx.x * blockDim.x + threadIdx.x;
  size_t total = (size_t)G4 * Hh;
  if (i >= total) return;
  int c = (int)(i % Hh);
  int n = (int)(i / Hh);
  Wx[i] = f2bf(Wih[(size_t)n * ih_ld + c]);
}

__global__ void pack_wt_kernel(const float* __restrict__ W, ushort_t* __restrict__ Wp) {
  // W [100][768] -> Wp [112][768] zero-padded rows
  size_t i = (size_t)blockIdx.x * blockDim.x + threadIdx.x;
  size_t total = (size_t)112 * Hh;
  if (i >= total) return;
  int c = (int)(i % Hh);
  int n = (int)(i / Hh);
  Wp[i] = (n < TAGn) ? f2bf(W[(size_t)n * Hh + c]) : (ushort_t)0;
}

__global__ void pack_bias_kernel(const float* __restrict__ bih, const float* __restrict__ bhh,
                                 float* __restrict__ bias) {
  int i = blockIdx.x * blockDim.x + threadIdx.x;
  if (i < G4) bias[i] = bih[i] + bhh[i];
}

__global__ void reset_tags_kernel(ushort_t* __restrict__ r) {
  int i = blockIdx.x * blockDim.x + threadIdx.x;   // 64 blocks * 256
  int b = i >> 8, c = i & 255;
  if (b < Bb) r[(size_t)b * RLD + Hh + c] = 0;     // clears tag + att slots
}

// ------------------------------------------------- input-GEMM precompute (off critical path)
// Gx[(s*64+b), 0..3071] = bias[n] + Xbf[s,b,:] @ Wx[n,:]^T    (bf16 WMMA, f32 acc)
__global__ void __launch_bounds__(TPB) gx_kernel(const ushort_t* __restrict__ Xbf,
                                                 const ushort_t* __restrict__ Wx,
                                                 const float* __restrict__ bias,
                                                 float* __restrict__ Gx) {
  const int lane = threadIdx.x & 31;
  const int wv = blockIdx.x * (TPB >> 5) + (threadIdx.x >> 5);
  const int nw = gridDim.x * (TPB >> 5);
  const int mTiles = (Ss * Bb) / 16;   // 2048
  const int nTiles = G4 / 16;          // 192
  for (int t = wv; t < mTiles * nTiles; t += nw) {
    int mt = t % mTiles, nt = t / mTiles;
    int m0 = mt << 4, n0 = nt << 4;
    int n = n0 + (lane & 15);
    float bv = bias[n];
    v8f acc;
#pragma unroll
    for (int v = 0; v < 8; ++v) acc[v] = bv;
    for (int k0 = 0; k0 < Hh; k0 += 32)
      acc = wmma_bf16(load_a_frag(Xbf, Hh, m0, k0, lane),
                      load_b_frag(Wx, Hh, n0, k0, lane), acc);
    int mbase = m0 + ((lane >> 4) << 3);
#pragma unroll
    for (int v = 0; v < 8; ++v)
      Gx[(size_t)(mbase + v) * G4 + n] = acc[v];
  }
}

// ---------------------------------------------------------------- recurrent phases
// gates GEMM: B fragments come from this wave's LDS-resident weight slice
__device__ __forceinline__ void gates_phase(int s, int wv, int lane,
    const ushort_t* Xbf, const ushort_t* Wx, const float* Gx, int use_gx,
    const ushort_t* r, const ushort_t* WrLds, int Kr, const float* bias,
    float* gbuf) {
  int m0 = (wv & 3) << 4;
  int n0 = (wv >> 2) << 4;
  int n  = n0 + (lane & 15);
  int mbase = m0 + ((lane >> 4) << 3);
  v8f acc;
  if (use_gx) {
    const float* g = Gx + ((size_t)s * Bb + mbase) * G4 + n;
#pragma unroll
    for (int v = 0; v < 8; ++v) acc[v] = g[(size_t)v * G4];
  } else {
    float bv = bias[n];
#pragma unroll
    for (int v = 0; v < 8; ++v) acc[v] = bv;
    const ushort_t* Xs = Xbf + (size_t)s * Bb * Hh;
    for (int k0 = 0; k0 < Hh; k0 += 32)
      acc = wmma_bf16(load_a_frag(Xs, Hh, m0, k0, lane),
                      load_b_frag(Wx, Hh, n0, k0, lane), acc);
  }
  for (int k0 = 0; k0 < Kr; k0 += 32)
    acc = wmma_bf16(load_a_frag(r, RLD, m0, k0, lane),
                    load_b_frag(WrLds, Kr, /*n0=*/0, k0, lane), acc);
  float* gp = gbuf + (size_t)mbase * G4 + n;
#pragma unroll
  for (int v = 0; v < 8; ++v) gp[(size_t)v * G4] = acc[v];
}

__device__ __forceinline__ void cell_phase(int gt, int nthr,
    const float* gbuf, float* cbuf, ushort_t* r) {
  for (int e = gt; e < Bb * Hh; e += nthr) {
    int b = e / Hh, u = e - b * Hh;
    const float* g = gbuf + (size_t)b * G4;
    float ig = sigmoidf_(g[u]);
    float fg = sigmoidf_(g[Hh + u]);
    float gg = tanhf(g[2 * Hh + u]);
    float og = sigmoidf_(g[3 * Hh + u]);
    float c2 = fg * cbuf[e] + ig * gg;
    cbuf[e] = c2;
    r[(size_t)b * RLD + u] = f2bf(og * tanhf(c2));
  }
}

__device__ __forceinline__ void tags_phase(int s, int wv, int lane,
    const ushort_t* Wp, const float* bvec, float* Tout, ushort_t* r) {
  if (wv >= 28) return;                    // 4 x 7 tiles cover 64 x 112
  int m0 = (wv & 3) << 4;
  int n0 = (wv >> 2) << 4;
  int t  = n0 + (lane & 15);
  float bv = (t < TAGn) ? bvec[t] : 0.0f;
  v8f acc;
#pragma unroll
  for (int v = 0; v < 8; ++v) acc[v] = bv;
  for (int k0 = 0; k0 < Hh; k0 += 32)      // A = h part of r (cols 0..767)
    acc = wmma_bf16(load_a_frag(r, RLD, m0, k0, lane),
                    load_b_frag(Wp, Hh, n0, k0, lane), acc);
  if (t < TAGn) {
    int mbase = m0 + ((lane >> 4) << 3);
#pragma unroll
    for (int v = 0; v < 8; ++v) {
      int b = mbase + v;
      float val = acc[v];
      Tout[((size_t)b * Ss + s) * TAGn + t] = val;
      r[(size_t)b * RLD + Hh + t] = f2bf(val);
    }
  }
}

// ---------------------------------------------------------------- stage 1 (persistent)
__global__ void __launch_bounds__(TPB) stage1_kernel(
    const ushort_t* __restrict__ Xbf, const ushort_t* __restrict__ Wx1,
    const float* __restrict__ Gx, int use_gx,
    const ushort_t* __restrict__ Wr1, const float* __restrict__ bias1,
    const ushort_t* __restrict__ W1p, const float* __restrict__ b1,
    ushort_t* r, float* cbuf, float* gbuf, float* T1, unsigned* cnt) {
  extern __shared__ ushort_t lds_w[];      // 8 waves * 16 rows * KR1 bf16 = 224 KB
  const int tid = threadIdx.x, lane = tid & 31;
  const int wv = (blockIdx.x << 3) | (tid >> 5);
  const int gt = blockIdx.x * TPB + tid;
  const unsigned nb = gridDim.x;
  const int nthr = gridDim.x * TPB;
  ushort_t* myW = lds_w + (size_t)(tid >> 5) * 16 * KR1;
  stage_wr_to_lds(myW, Wr1, KR1, (wv >> 2) << 4, lane);
  __syncthreads();
  unsigned gen = 0;
  for (int s = 0; s < Ss; ++s) {
    gates_phase(s, wv, lane, Xbf, Wx1, Gx, use_gx, r, myW, KR1, bias1, gbuf);
    grid_bar(cnt, ++gen, nb);
    cell_phase(gt, nthr, gbuf, cbuf, r);
    grid_bar(cnt, ++gen, nb);
    tags_phase(s, wv, lane, W1p, b1, T1, r);
    grid_bar(cnt, ++gen, nb);
  }
}

// ---------------------------------------------------------------- stage 2 (persistent, + attention)
__global__ void __launch_bounds__(TPB) stage2_kernel(
    const ushort_t* __restrict__ Xbf, const ushort_t* __restrict__ Wx2,
    const float* __restrict__ Gx, int use_gx,
    const ushort_t* __restrict__ Wr2, const float* __restrict__ bias2,
    const ushort_t* __restrict__ W2p, const float* __restrict__ b2,
    const float* __restrict__ Wa, const float* __restrict__ ba,
    const float* __restrict__ T1, float* T2,
    ushort_t* r, float* cbuf, float* gbuf, unsigned* cnt) {
  extern __shared__ ushort_t lds_w[];      // 8 waves * 16 rows * KR2 bf16 = 256 KB
  __shared__ float s_tidx[112];
  __shared__ float s_z[Ss];
  __shared__ float s_red[TPB];
  const int tid = threadIdx.x, lane = tid & 31;
  const int wv = (blockIdx.x << 3) | (tid >> 5);
  const int gt = blockIdx.x * TPB + tid;
  const unsigned nb = gridDim.x;
  const int nthr = gridDim.x * TPB;
  ushort_t* myW = lds_w + (size_t)(tid >> 5) * 16 * KR2;
  stage_wr_to_lds(myW, Wr2, KR2, (wv >> 2) << 4, lane);
  __syncthreads();
  unsigned gen = 0;
  for (int s = 0; s < Ss; ++s) {
    // ---- attention: one workgroup per batch row (blocks 0..63) ----
    if (blockIdx.x < Bb) {
      int b = blockIdx.x;
      const float* t1row = T1 + ((size_t)b * Ss + s) * TAGn;
      if (tid < 112) {
        float a0 = 0.0f;
        if (tid < TAGn) {
          a0 = ba[tid];
          const float* war = Wa + (size_t)tid * TAGn;
          for (int u = 0; u < TAGn; ++u) a0 += war[u] * t1row[u];
          a0 = tanhf(a0);
        }
        s_tidx[tid] = a0;
      }
      __syncthreads();
      float lmax = -3.0e38f;
      for (int sa = tid; sa < Ss; sa += TPB) {
        const float* row = T1 + ((size_t)b * Ss + sa) * TAGn;
        float z = 0.0f;
        for (int u = 0; u < TAGn; ++u) z += s_tidx[u] * row[u];
        s_z[sa] = z;
        lmax = fmaxf(lmax, z);
      }
      s_red[tid] = lmax; __syncthreads();
      for (int off = TPB >> 1; off > 0; off >>= 1) {
        if (tid < off) s_red[tid] = fmaxf(s_red[tid], s_red[tid + off]);
        __syncthreads();
      }
      float mx = s_red[0]; __syncthreads();
      float lsum = 0.0f;
      for (int sa = tid; sa < Ss; sa += TPB) {
        float e = expf(s_z[sa] - mx);
        s_z[sa] = e;
        lsum += e;
      }
      s_red[tid] = lsum; __syncthreads();
      for (int off = TPB >> 1; off > 0; off >>= 1) {
        if (tid < off) s_red[tid] += s_red[tid + off];
        __syncthreads();
      }
      float inv = 1.0f / s_red[0];
      __syncthreads();
      if (tid < TAGn) {
        float acc = 0.0f;
        for (int sa = 0; sa < Ss; ++sa)
          acc += s_z[sa] * T1[((size_t)b * Ss + sa) * TAGn + tid];
        r[(size_t)b * RLD + Hh + TAGP + tid] = f2bf(acc * inv);  // att slot
      }
    }
    grid_bar(cnt, ++gen, nb);
    gates_phase(s, wv, lane, Xbf, Wx2, Gx, use_gx, r, myW, KR2, bias2, gbuf);
    grid_bar(cnt, ++gen, nb);
    cell_phase(gt, nthr, gbuf, cbuf, r);
    grid_bar(cnt, ++gen, nb);
    tags_phase(s, wv, lane, W2p, b2, T2, r);
    grid_bar(cnt, ++gen, nb);
  }
}

// ---------------------------------------------------------------- final projection
__global__ void final_kernel(const float* __restrict__ T2, const float* __restrict__ W3,
                             const float* __restrict__ b3, float* __restrict__ out) {
  size_t i = (size_t)blockIdx.x * blockDim.x + threadIdx.x;
  size_t total = (size_t)Bb * Ss * ROLEn;
  if (i >= total) return;
  int rr = (int)(i % ROLEn);
  size_t bs = i / ROLEn;
  float acc = b3[rr];
  const float* t = T2 + bs * TAGn;
  const float* w = W3 + (size_t)rr * TAGn;
  for (int u = 0; u < TAGn; ++u) acc += t[u] * w[u];
  out[i] = acc;
}

// ---------------------------------------------------------------- host launcher
extern "C" void kernel_launch(void* const* d_in, const int* in_sizes, int n_in,
                              void* d_out, int out_size, void* d_ws, size_t ws_size,
                              hipStream_t stream) {
  (void)in_sizes; (void)n_in; (void)out_size;
  const float* X    = (const float*)d_in[0];
  const float* Wih1 = (const float*)d_in[1];
  const float* Whh1 = (const float*)d_in[2];
  const float* bih1 = (const float*)d_in[3];
  const float* bhh1 = (const float*)d_in[4];
  const float* Wih2 = (const float*)d_in[5];
  const float* Whh2 = (const float*)d_in[6];
  const float* bih2 = (const float*)d_in[7];
  const float* bhh2 = (const float*)d_in[8];
  const float* W1   = (const float*)d_in[9];
  const float* b1   = (const float*)d_in[10];
  const float* W2   = (const float*)d_in[11];
  const float* b2   = (const float*)d_in[12];
  const float* Wa   = (const float*)d_in[13];
  const float* ba   = (const float*)d_in[14];
  const float* W3   = (const float*)d_in[15];
  const float* b3   = (const float*)d_in[16];
  float* out = (float*)d_out;

  char* ws = (char*)d_ws;
  size_t off = 0;
  auto alloc = [&](size_t bytes) -> char* {
    char* p = ws + off;
    off = (off + bytes + 255) & ~(size_t)255;
    return p;
  };
  unsigned* cnt  = (unsigned*)alloc(256);
  ushort_t* rbuf = (ushort_t*)alloc((size_t)Bb * RLD * 2);
  float*  cbuf   = (float*)   alloc((size_t)Bb * Hh * 4);
  float*  gbuf   = (float*)   alloc((size_t)Bb * G4 * 4);
  ushort_t* Xbf  = (ushort_t*)alloc((size_t)Ss * Bb * Hh * 2);
  ushort_t* Wr1  = (ushort_t*)alloc((size_t)G4 * KR1 * 2);
  ushort_t* Wr2  = (ushort_t*)alloc((size_t)G4 * KR2 * 2);
  ushort_t* Wx1  = (ushort_t*)alloc((size_t)G4 * Hh * 2);
  ushort_t* Wx2  = (ushort_t*)alloc((size_t)G4 * Hh * 2);
  ushort_t* W1p  = (ushort_t*)alloc((size_t)112 * Hh * 2);
  ushort_t* W2p  = (ushort_t*)alloc((size_t)112 * Hh * 2);
  float*  bias1  = (float*)   alloc((size_t)G4 * 4);
  float*  bias2  = (float*)   alloc((size_t)G4 * 4);
  float*  T1     = (float*)   alloc((size_t)Bb * Ss * TAGn * 4);
  float*  T2     = (float*)   alloc((size_t)Bb * Ss * TAGn * 4);
  size_t base_need = off;
  size_t gx_bytes = (size_t)Ss * Bb * G4 * 4;       // 384 MB
  int use_gx = (ws_size >= base_need + gx_bytes + 256) ? 1 : 0;
  float* Gx = use_gx ? (float*)alloc(gx_bytes) : nullptr;

  // per-call init (graph-capture safe)
  hipMemsetAsync(cnt, 0, 256, stream);
  hipMemsetAsync(rbuf, 0, (size_t)Bb * RLD * 2, stream);
  hipMemsetAsync(cbuf, 0, (size_t)Bb * Hh * 4, stream);

  auto nblocks = [](size_t n) { return (unsigned)((n + TPB - 1) / TPB); };

  pack_x_kernel<<<nblocks((size_t)Ss * Bb * Hh), TPB, 0, stream>>>(X, Xbf);
  pack_wr_kernel<<<nblocks((size_t)G4 * KR1), TPB, 0, stream>>>(Whh1, Wih1, Wr1, KR1, Hh + TAGn);
  pack_wr_kernel<<<nblocks((size_t)G4 * KR2), TPB, 0, stream>>>(Whh2, Wih2, Wr2, KR2, Hh + 2 * TAGn);
  pack_wx_kernel<<<nblocks((size_t)G4 * Hh), TPB, 0, stream>>>(Wih1, Wx1, Hh + TAGn);
  pack_wx_kernel<<<nblocks((size_t)G4 * Hh), TPB, 0, stream>>>(Wih2, Wx2, Hh + 2 * TAGn);
  pack_wt_kernel<<<nblocks((size_t)112 * Hh), TPB, 0, stream>>>(W1, W1p);
  pack_wt_kernel<<<nblocks((size_t)112 * Hh), TPB, 0, stream>>>(W2, W2p);
  pack_bias_kernel<<<nblocks(G4), TPB, 0, stream>>>(bih1, bhh1, bias1);
  pack_bias_kernel<<<nblocks(G4), TPB, 0, stream>>>(bih2, bhh2, bias2);

  const size_t lds1 = (size_t)8 * 16 * KR1 * 2;   // 229376 B
  const size_t lds2 = (size_t)8 * 16 * KR2 * 2;   // 262144 B

  if (use_gx)
    gx_kernel<<<1536, TPB, 0, stream>>>(Xbf, Wx1, bias1, Gx);
  stage1_kernel<<<NBLK, TPB, lds1, stream>>>(Xbf, Wx1, Gx, use_gx, Wr1, bias1,
                                             W1p, b1, rbuf, cbuf, gbuf, T1, cnt);

  hipMemsetAsync(cnt, 0, 256, stream);
  reset_tags_kernel<<<Bb, TPB, 0, stream>>>(rbuf);   // tags0 = 0 for stage 2 (h, c carry over)
  if (use_gx)
    gx_kernel<<<1536, TPB, 0, stream>>>(Xbf, Wx2, bias2, Gx);
  stage2_kernel<<<NBLK, TPB, lds2, stream>>>(Xbf, Wx2, Gx, use_gx, Wr2, bias2,
                                             W2p, b2, Wa, ba, T1, T2,
                                             rbuf, cbuf, gbuf, cnt);

  final_kernel<<<nblocks((size_t)Bb * Ss * ROLEn), TPB, 0, stream>>>(T2, W3, b3, out);
}
